// Borzoi_27925877358946
// MI455X (gfx1250) — matrix operand order repaired and checked by reading, not compiled, not run on hardware
//
#include <hip/hip_runtime.h>
#include <hip/hip_fp16.h>

typedef _Float16 h16;
typedef __attribute__((ext_vector_type(16))) _Float16 v16h;
typedef __attribute__((ext_vector_type(8)))  _Float16 v8h;
typedef __attribute__((ext_vector_type(8)))  float    v8f;

#define LDSROW 48   // LDS row stride in halfs: 64B data + 32B pad = 96B (32B-aligned)

__device__ __forceinline__ float gelu_f(float x) {
    return 0.5f * x * (1.0f + tanhf(0.7978845608f * (x + 0.044715f * x * x * x)));
}
__device__ __forceinline__ v8h v8h_zero() {
    v8h z;
    #pragma unroll
    for (int i = 0; i < 8; i++) z[i] = (h16)0.f;
    return z;
}

// ---------------------------------------------------------------------------
// Tensor Data Mover tile load: 2D tile [128 rows x 32 halfs] -> LDS with
// 48-half row stride via D# padding (pad every 16 dwords by 8 dwords).
// Rows beyond `rows_valid` zero-fill (ISA: OOB reads return zero).
// 6-arg builtin: (uint32x4 g0, int32x8 g1, int32x4 g2, int32x4 g3, int32x8, i32 cpol)
// ---------------------------------------------------------------------------
#if defined(__gfx1250__) && __has_builtin(__builtin_amdgcn_tensor_load_to_lds) && __has_builtin(__builtin_amdgcn_s_wait_tensorcnt)
#define USE_TDM 1
typedef unsigned __attribute__((ext_vector_type(4))) tdm_u4;
typedef int      __attribute__((ext_vector_type(8))) tdm_i8;
typedef int      __attribute__((ext_vector_type(4))) tdm_i4;

__device__ __forceinline__ void tdm_load_tile_2d(const void* gaddr, unsigned lds_byte_addr,
                                                 unsigned rows_valid, unsigned ld_elems) {
    unsigned long ga = (unsigned long)gaddr;
    tdm_u4 g0;
    g0[0] = 1u;                                            // count=1, user descriptor
    g0[1] = lds_byte_addr;                                 // lds_addr
    g0[2] = (unsigned)ga;                                  // global_addr[31:0]
    g0[3] = (unsigned)((ga >> 32) & 0x01ffffffu) | (2u << 30);  // addr[56:32] | type=2
    tdm_i8 g1;
    g1[0] = (int)((1u << 16) | (1u << 20) | (3u << 22) | (7u << 25));
    //            data=2B  |  pad_enable | every 16 dw  | pad 8 dw   -> 96B LDS rows
    g1[1] = (int)(32u << 16);                              // tensor_dim0 = 32 (lo16)
    g1[2] = (int)((rows_valid & 0xffffu) << 16);           // dim0 hi=0 | tensor_dim1 lo16
    g1[3] = (int)((rows_valid >> 16) & 0xffffu) | (int)(32u << 16);  // dim1 hi | tile_dim0=32
    g1[4] = 128;                                           // tile_dim1=128, tile_dim2=0
    g1[5] = (int)ld_elems;                                 // tensor_dim0_stride[31:0]
    g1[6] = 0;                                             // stride hi | dim1_stride lo
    g1[7] = 0;
    tdm_i4 gz; gz[0] = 0; gz[1] = 0; gz[2] = 0; gz[3] = 0;
    tdm_i8 gx;
    #pragma unroll
    for (int i = 0; i < 8; i++) gx[i] = 0;
    __builtin_amdgcn_tensor_load_to_lds(g0, g1, gz, gz, gx, 0);
}
#else
#define USE_TDM 0
#endif

// ---------------------------------------------------------------------------
// Generic WMMA GEMM:  Out[M][N] = act(A[M][K] * W[N][K]^T + bias)
//   CONV=1 : implicit im2col over K = CIN*KW taps, SAME padding (CIN%32==0)
//   ACT    : 0 none, 1 gelu(tanh), 2 relu
//   POOL=1 : fused maxpool2 along M (row pairs live in one lane)
//   OUTMODE: 0 fp16 [M][ldc], 1 fp32 [M][ldc], 2 fp32 transposed [N][ldc],
//            3 fp32 [M][ldc] += (residual), 4 fp16 transposed [N][ldc]
// 256 threads = 8 waves (wave32); tile 128x128x32; wave = 64x32 (8 WMMAs/step).
// Double-buffered LDS. TDM moves B always, A when its tile is rectangular
// (always for !CONV; for CONV whenever pos0 = m0+t-PAD >= 0 — the top-edge
// tiles fall back to register staging with explicit zero-fill).
// ---------------------------------------------------------------------------
template<int CONV, int ACT, int POOL, int OUTMODE>
__global__ __launch_bounds__(256)
void gemm_wmma(const h16* __restrict__ A, int lda,
               const h16* __restrict__ W, int ldb,
               const float* __restrict__ bias,
               void* __restrict__ Out, int ldc,
               int M, int N, int K,
               int CIN, int KW, int PAD)
{
    __shared__ __align__(32) h16 As[2][128 * LDSROW];
    __shared__ __align__(32) h16 Bs[2][128 * LDSROW];

    const int tid  = threadIdx.x;
    const int lane = tid & 31;
    const int wave = tid >> 5;
    const int wm   = wave >> 2;            // 0..1  (rows of 64)
    const int wn   = wave & 3;             // 0..3  (cols of 32)
    const int m0   = blockIdx.y * 128;
    const int n0   = blockIdx.x * 128;

    v8f acc[4][2];
    #pragma unroll
    for (int i = 0; i < 4; i++)
        #pragma unroll
        for (int j = 0; j < 2; j++)
            #pragma unroll
            for (int e = 0; e < 8; e++) acc[i][j][e] = 0.f;

    // -- register-staged loaders (CONV top-edge tiles; everything in fallback)
    v8h ra[2], rb[2];
    auto load_a = [&](int kt) {
        #pragma unroll
        for (int q = 0; q < 2; q++) {
            int c = tid + q * 256, m = c >> 2, kk = (c & 3) * 8;
            bool ok; const h16* src;
            if (CONV) {
                int gk = kt + kk;
                int t  = gk / CIN;
                int ci = gk - t * CIN;
                long pos = (long)(m0 + m) + t - PAD;
                ok  = ((m0 + m) < M) && pos >= 0 && pos < M;
                src = A + pos * (long)CIN + ci;
            } else {
                ok  = (m0 + m) < M;
                src = A + (long)(m0 + m) * lda + (kt + kk);
            }
            ra[q] = ok ? *(const v8h*)src : v8h_zero();
        }
    };
    auto store_a = [&](int buf) {
        #pragma unroll
        for (int q = 0; q < 2; q++) {
            int c = tid + q * 256, m = c >> 2, kk = (c & 3) * 8;
            *(v8h*)(As[buf] + m * LDSROW + kk) = ra[q];
        }
    };
    auto load_b = [&](int kt) {
        #pragma unroll
        for (int q = 0; q < 2; q++) {
            int c = tid + q * 256, n = c >> 2, kk = (c & 3) * 8;
            rb[q] = ((n0 + n) < N) ? *(const v8h*)(W + (long)(n0 + n) * ldb + kt + kk)
                                   : v8h_zero();
        }
    };
    auto store_b = [&](int buf) {
        #pragma unroll
        for (int q = 0; q < 2; q++) {
            int c = tid + q * 256, n = c >> 2, kk = (c & 3) * 8;
            *(v8h*)(Bs[buf] + n * LDSROW + kk) = rb[q];
        }
    };

    const int nk = K >> 5;

#if USE_TDM
    const bool w0 = (tid < 32);            // wave 0 drives the TDM
    // issue TDM loads (A conditionally) for k-tile `kt` into buffer `buf`;
    // returns true if the A tile needs the manual (register-staged) path.
    auto issue_tile = [&](int kt, int buf) -> bool {
        bool a_manual = false;
        long apos = 0; int aci = 0;
        if (CONV) {
            int t = kt / CIN;
            aci   = kt - t * CIN;
            apos  = (long)m0 + t - PAD;
            a_manual = (apos < 0);         // top edge: needs zero-fill before base
        }
        if (w0) {
            tdm_load_tile_2d(W + (long)n0 * ldb + kt,
                             (unsigned)(unsigned long)(const void*)Bs[buf],
                             (unsigned)(N - n0), (unsigned)ldb);
            if (!CONV)
                tdm_load_tile_2d(A + (long)m0 * lda + kt,
                                 (unsigned)(unsigned long)(const void*)As[buf],
                                 (unsigned)(M - m0), (unsigned)lda);
            else if (!a_manual)
                tdm_load_tile_2d(A + apos * (long)CIN + aci,
                                 (unsigned)(unsigned long)(const void*)As[buf],
                                 (unsigned)(M - (int)apos), (unsigned)CIN);
        }
        return a_manual;
    };

    bool a_man = issue_tile(0, 0);
    if (CONV && a_man) load_a(0);
#else
    load_a(0);
    load_b(0);
#endif

    for (int it = 0; it < nk; ++it) {
        const int buf = it & 1;
        const int kt  = it << 5;
#if USE_TDM
        if (CONV && a_man) store_a(buf);
        if (w0) __builtin_amdgcn_s_wait_tensorcnt(0);
        __syncthreads();
        if (it + 1 < nk) {
            bool a_man_n = issue_tile(kt + 32, buf ^ 1);
            if (CONV && a_man_n) load_a(kt + 32);
            a_man = a_man_n;
        }
#else
        store_a(buf);
        store_b(buf);
        __syncthreads();
        if (it + 1 < nk) { load_a(kt + 32); load_b(kt + 32); }
#endif
        // ---- fragments in contiguous k order (same for A and B); 8 WMMAs per wave per step ----
        const int rsel = lane >> 4;
        const int cidx = lane & 15;
        const int ka0  = rsel * 16;        // A frag: K {ka0..ka0+15} contiguous
        const int kb0  = rsel * 16;        // B frag: K {kb0..kb0+15} contiguous

        v16h bf[2];
        #pragma unroll
        for (int nn = 0; nn < 2; nn++)
            bf[nn] = *(const v16h*)(Bs[buf] + (wn * 32 + nn * 16 + cidx) * LDSROW + kb0);

        #pragma unroll
        for (int mm = 0; mm < 4; mm++) {
            const h16* ap = As[buf] + (wm * 64 + mm * 16 + cidx) * LDSROW;
            v8h lo = *(const v8h*)(ap + ka0);
            v8h hi = *(const v8h*)(ap + ka0 + 8);
            v16h af;
            #pragma unroll
            for (int e = 0; e < 8; e++) { af[e] = lo[e]; af[e + 8] = hi[e]; }
            #pragma unroll
            for (int nn = 0; nn < 2; nn++)
                acc[mm][nn] = __builtin_amdgcn_wmma_f32_16x16x32_f16(
                    false, af, false, bf[nn], (short)0, acc[mm][nn], false, false);
        }
    }

    // ---- epilogue ----
    #pragma unroll
    for (int mm = 0; mm < 4; mm++) {
        #pragma unroll
        for (int nn = 0; nn < 2; nn++) {
            int cg    = n0 + wn * 32 + nn * 16 + (lane & 15);
            int rbase = m0 + wm * 64 + mm * 16 + (lane >> 4) * 8;
            if (cg >= N) continue;
            float vals[8];
            #pragma unroll
            for (int e = 0; e < 8; e++) {
                float v = acc[mm][nn][e];
                if (bias) v += bias[cg];
                if (ACT == 1) v = gelu_f(v);
                if (ACT == 2) v = fmaxf(v, 0.f);
                vals[e] = v;
            }
            if (POOL) {
                h16* O = (h16*)Out;
                #pragma unroll
                for (int p = 0; p < 4; p++) {
                    if (rbase + 2 * p < M) {
                        int pr = (rbase >> 1) + p;
                        O[(long)pr * ldc + cg] = (h16)fmaxf(vals[2 * p], vals[2 * p + 1]);
                    }
                }
            } else {
                #pragma unroll
                for (int e = 0; e < 8; e++) {
                    int r = rbase + e;
                    if (r >= M) continue;
                    if (OUTMODE == 0)      ((h16*)Out)[(long)r * ldc + cg] = (h16)vals[e];
                    else if (OUTMODE == 1) ((float*)Out)[(long)r * ldc + cg] = vals[e];
                    else if (OUTMODE == 2) ((float*)Out)[(long)cg * ldc + r] = vals[e];
                    else if (OUTMODE == 3) ((float*)Out)[(long)r * ldc + cg] += vals[e];
                    else if (OUTMODE == 4) ((h16*)Out)[(long)cg * ldc + r] = (h16)vals[e];
                }
            }
        }
    }
}

// ---------------------------------------------------------------------------
// Elementwise / helper kernels (grid-stride unless noted)
// ---------------------------------------------------------------------------
__global__ void k_cvt_f32_f16(const float* __restrict__ in, h16* __restrict__ out, long n) {
    for (long i = blockIdx.x * 256L + threadIdx.x; i < n; i += (long)gridDim.x * 256)
        out[i] = (h16)in[i];
}
__global__ void k_cvt_f16_f32(const h16* __restrict__ in, float* __restrict__ out, long n) {
    for (long i = blockIdx.x * 256L + threadIdx.x; i < n; i += (long)gridDim.x * 256)
        out[i] = (float)in[i];
}
// dense fp32 W[K][N] -> fp16 Wt[N][K]
__global__ void k_wt_dense(const float* __restrict__ w, h16* __restrict__ wt, int K, int N) {
    for (long i = blockIdx.x * 256L + threadIdx.x; i < (long)K * N; i += (long)gridDim.x * 256) {
        int n = (int)(i % N); int k = (int)(i / N);
        wt[(long)n * K + k] = (h16)w[i];
    }
}
// conv fp32 w[CO][CI][KW] -> fp16 Wt[CO][t*CI+ci]
__global__ void k_wt_conv(const float* __restrict__ w, h16* __restrict__ wt, int CO, int CI, int KW) {
    long tot = (long)CO * CI * KW;
    for (long i = blockIdx.x * 256L + threadIdx.x; i < tot; i += (long)gridDim.x * 256) {
        int co = (int)(i / ((long)CI * KW));
        int rem = (int)(i - (long)co * CI * KW);
        int ci = rem / KW, t = rem - ci * KW;
        wt[(long)co * CI * KW + (long)t * CI + ci] = (h16)w[i];
    }
}
// dna w[512][4][15] -> Wt[512][64], k = t*4+ci, K padded to 64
__global__ void k_wt_dna(const float* __restrict__ w, h16* __restrict__ wt) {
    int i = blockIdx.x * 256 + threadIdx.x;
    if (i < 512 * 64) {
        int co = i >> 6, k = i & 63, t = k >> 2, ci = k & 3;
        wt[i] = (t < 15) ? (h16)w[(co * 4 + ci) * 15 + t] : (h16)0.f;
    }
}
// im2col for dna conv: x fp32 [4][L] -> out fp16 [L][64]
__global__ void k_dna_im2col(const float* __restrict__ x, h16* __restrict__ out, long L) {
    long tot = L * 64;
    for (long i = blockIdx.x * 256L + threadIdx.x; i < tot; i += (long)gridDim.x * 256) {
        long m = i >> 6; int k = (int)(i & 63); int t = k >> 2, ci = k & 3;
        long pos = m + t - 7;
        float v = (t < 15 && pos >= 0 && pos < L) ? x[(long)ci * L + pos] : 0.f;
        out[i] = (h16)v;
    }
}
__global__ void k_bngelu(const h16* __restrict__ x, const float* __restrict__ g,
                         const float* __restrict__ b, h16* __restrict__ y, long n, int C) {
    for (long i = blockIdx.x * 256L + threadIdx.x; i < n; i += (long)gridDim.x * 256) {
        int c = (int)(i % C);
        float v = (float)x[i] * (g[c] * 0.99950037f) + b[c];   // 1/sqrt(1+1e-3)
        y[i] = (h16)gelu_f(v);
    }
}
__global__ void k_maxpool(const h16* __restrict__ x, h16* __restrict__ y, long n, int C) {
    for (long i = blockIdx.x * 256L + threadIdx.x; i < n; i += (long)gridDim.x * 256) {
        long p = i / C; int c = (int)(i % C);
        float a = (float)x[(2 * p) * C + c], bb = (float)x[(2 * p + 1) * C + c];
        y[i] = (h16)fmaxf(a, bb);
    }
}
__global__ void k_repeat_add(const h16* __restrict__ u, const h16* __restrict__ s,
                             h16* __restrict__ y, long n, int C) {
    for (long i = blockIdx.x * 256L + threadIdx.x; i < n; i += (long)gridDim.x * 256) {
        long p = i / C; int c = (int)(i % C);
        y[i] = (h16)((float)u[(p >> 1) * C + c] + (float)s[i]);
    }
}
__global__ void k_dwconv3(const h16* __restrict__ x, const float* __restrict__ w,
                          h16* __restrict__ y, long M, int C) {
    long n = M * C;
    for (long i = blockIdx.x * 256L + threadIdx.x; i < n; i += (long)gridDim.x * 256) {
        long p = i / C; int c = (int)(i % C);
        float acc = 0.f;
        #pragma unroll
        for (int t = 0; t < 3; t++) {
            long pp = p + t - 1;
            if (pp >= 0 && pp < M) acc += w[c * 3 + t] * (float)x[pp * C + c];
        }
        y[i] = (h16)acc;
    }
}
// LayerNorm over C, one block per row; fp32 in -> fp16 out
__global__ __launch_bounds__(256) void k_ln(const float* __restrict__ X, const float* __restrict__ g,
                                            const float* __restrict__ b, h16* __restrict__ Y, int C) {
    int row = blockIdx.x;
    const float* x = X + (long)row * C;
    __shared__ float red[256];
    float s = 0.f;
    for (int c = threadIdx.x; c < C; c += 256) s += x[c];
    red[threadIdx.x] = s; __syncthreads();
    for (int st = 128; st > 0; st >>= 1) { if (threadIdx.x < st) red[threadIdx.x] += red[threadIdx.x + st]; __syncthreads(); }
    float mean = red[0] / C; __syncthreads();
    float v = 0.f;
    for (int c = threadIdx.x; c < C; c += 256) { float d = x[c] - mean; v += d * d; }
    red[threadIdx.x] = v; __syncthreads();
    for (int st = 128; st > 0; st >>= 1) { if (threadIdx.x < st) red[threadIdx.x] += red[threadIdx.x + st]; __syncthreads(); }
    float rstd = rsqrtf(red[0] / C + 1e-3f);
    h16* y = Y + (long)row * C;
    for (int c = threadIdx.x; c < C; c += 256) y[c] = (h16)((x[c] - mean) * rstd * g[c] + b[c]);
}
// pos-embed features pe[8191][32]
__global__ void k_pe(h16* __restrict__ PE, int NT) {
    long i = blockIdx.x * 256L + threadIdx.x;
    long tot = 2L * NT - 1;
    if (i >= tot) return;
    float d = (float)(i - (NT - 1));
    float rate = expf(logf((float)NT + 1.0f) / 16.0f);
    h16* row = PE + i * 32;
    float sg = (d > 0.f) ? 1.f : ((d < 0.f) ? -1.f : 0.f);
    float cw = 1.0f;
    for (int j = 0; j < 16; j++) {
        cw *= rate;
        float e = ((cw - 1.0f) > fabsf(d)) ? 1.0f : 0.0f;
        row[j] = (h16)e;
        row[16 + j] = (h16)(sg * e);
    }
}
// qc = q*scale + rcb ; qp = q*scale + rpb   (row stride 512 = heads*dk)
__global__ void k_qcqp(const float* __restrict__ qf, const float* __restrict__ rcb,
                       const float* __restrict__ rpb, h16* __restrict__ qc,
                       h16* __restrict__ qp, long n) {
    for (long i = blockIdx.x * 256L + threadIdx.x; i < n; i += (long)gridDim.x * 256) {
        int c = (int)(i & 511);
        float q = qf[i] * 0.125f;
        qc[i] = (h16)(q + rcb[c]);
        qp[i] = (h16)(q + rpb[c]);
    }
}
// fused relative-shift gather + softmax: P[i][j] = softmax_j(S[i][j] + R[i][j-i+N-1])
__global__ __launch_bounds__(256) void k_softmax_rel(const float* __restrict__ S,
                                                     const float* __restrict__ R,
                                                     h16* __restrict__ P, int N, int ldR) {
    int i = blockIdx.x;
    const float* srow = S + (long)i * N;
    const float* rrow = R + (long)i * ldR;
    float loc[16];
    float mx = -1e30f;
    #pragma unroll
    for (int t = 0; t < 16; t++) {
        int j = threadIdx.x + t * 256;
        float v = srow[j] + rrow[j - i + (N - 1)];
        loc[t] = v; mx = fmaxf(mx, v);
    }
    __shared__ float red[256];
    red[threadIdx.x] = mx; __syncthreads();
    for (int st = 128; st > 0; st >>= 1) { if (threadIdx.x < st) red[threadIdx.x] = fmaxf(red[threadIdx.x], red[threadIdx.x + st]); __syncthreads(); }
    mx = red[0]; __syncthreads();
    float sm = 0.f;
    #pragma unroll
    for (int t = 0; t < 16; t++) { loc[t] = expf(loc[t] - mx); sm += loc[t]; }
    red[threadIdx.x] = sm; __syncthreads();
    for (int st = 128; st > 0; st >>= 1) { if (threadIdx.x < st) red[threadIdx.x] += red[threadIdx.x + st]; __syncthreads(); }
    float inv = 1.0f / red[0];
    h16* prow = P + (long)i * N;
    #pragma unroll
    for (int t = 0; t < 16; t++) prow[threadIdx.x + t * 256] = (h16)(loc[t] * inv);
}

// ---------------------------------------------------------------------------
// Host orchestration
// ---------------------------------------------------------------------------
static inline dim3 gemm_grid(long M, long N) {
    return dim3((unsigned)((N + 127) / 128), (unsigned)((M + 127) / 128));
}
static inline dim3 ew_grid(long n) {
    long b = (n + 255) / 256;
    if (b > 1048576) b = 1048576;
    return dim3((unsigned)b);
}

extern "C" void kernel_launch(void* const* d_in, const int* in_sizes, int n_in,
                              void* d_out, int out_size, void* d_ws, size_t ws_size,
                              hipStream_t stream) {
    (void)in_sizes; (void)n_in; (void)out_size; (void)ws_size;

    const float* X     = (const float*)d_in[0];
    const float* dna_w = (const float*)d_in[1];
    const float* dna_b = (const float*)d_in[2];
    const int RT = 3;        // res_tower base: 4 arrays each {g,b,w,wb}
    const int UN = 23;       // unet1
    const int LB = 27;       // layers base: 16 arrays per layer
    const int HC0 = 155, HC1 = 159, UP1 = 163, SEP1 = 167, UP0 = 170, SEP0 = 174, FIN = 177;
    auto in_f = [&](int i) { return (const float*)d_in[i]; };

    // ---- workspace bump allocator ----
    size_t off = 0;
    auto alloc = [&](size_t bytes) -> char* {
        char* p = (char*)d_ws + off;
        off = (off + bytes + 255) & ~(size_t)255;
        return p;
    };
    const long L0 = 524288, L1 = 262144;
    h16* IM2  = (h16*)alloc(L0 * 64 * 2);                 // dna im2col
    h16* A0   = (h16*)alloc(L1 * 512 * 2);                // 256MB ping region
    h16* C1   = (h16*)alloc((long)131072 * 608 * 2);      // 152MB pong region
    h16* PRE  = (h16*)alloc(L1 * 512 * 2);                // 256MB pre-activation buf
    h16* CW   = (h16*)alloc((long)1536 * 6400 * 2);       // conv weight slot
    h16* WS   = (h16*)alloc((long)3072 * 1536 * 2);       // dense weight slot
    h16* XU1  = (h16*)alloc((long)8192 * 1536 * 2);
    h16* XU0C = (h16*)alloc((long)16384 * 1536 * 2);
    h16* XU1C = (h16*)alloc((long)8192 * 1536 * 2);
    h16* X4   = (h16*)alloc((long)4096 * 1536 * 2);
    float* T  = (float*)alloc((long)4096 * 1536 * 4);     // fp32 residual stream
    h16* Y    = (h16*)alloc((long)4096 * 1536 * 2);
    float* QF = (float*)alloc((long)4096 * 512 * 4);
    h16* QC   = (h16*)alloc((long)4096 * 512 * 2);
    h16* QP   = (h16*)alloc((long)4096 * 512 * 2);
    h16* KB   = (h16*)alloc((long)4096 * 512 * 2);
    h16* VT   = (h16*)alloc((long)1536 * 4096 * 2);
    h16* RELK = (h16*)alloc((long)8191 * 512 * 2);
    h16* PEB  = (h16*)alloc((long)8191 * 32 * 2);
    h16* ATTN = (h16*)alloc((long)4096 * 1536 * 2);
    h16* FF1  = (h16*)alloc((long)4096 * 3072 * 2);
    h16* XT   = (h16*)alloc((long)4096 * 1536 * 2);
    h16* U1B  = (h16*)alloc((long)4096 * 1536 * 2);
    h16* X8   = (h16*)alloc((long)8192 * 1536 * 2);
    h16* D8   = (h16*)alloc((long)8192 * 1536 * 2);
    h16* S8   = (h16*)alloc((long)8192 * 1536 * 2);
    h16* U8   = (h16*)alloc((long)8192 * 1536 * 2);
    h16* X16  = (h16*)alloc((long)16384 * 1536 * 2);
    h16* D16  = (h16*)alloc((long)16384 * 1536 * 2);
    h16* S16  = (h16*)alloc((long)16384 * 1536 * 2);
    // attention scratch overlays (regions free during transformer)
    float* Sbuf = (float*)A0;                                   // 64MB
    h16*   Pbuf = (h16*)((char*)A0 + (size_t)64 * 1024 * 1024); // 32MB
    float* Rbuf = (float*)PRE;                                  // 128MB (ld 8192)

    // ---- GEMM launchers ----
    auto g_f16 = [&](const h16* A, int lda, const h16* W, int ldb, const float* bias,
                     h16* O, int ldc, int M, int N, int K) {
        gemm_wmma<0,0,0,0><<<gemm_grid(M,N),256,0,stream>>>(A,lda,W,ldb,bias,(void*)O,ldc,M,N,K,0,1,0);
    };
    auto g_relu = [&](const h16* A, int lda, const h16* W, int ldb, const float* bias,
                      h16* O, int ldc, int M, int N, int K) {
        gemm_wmma<0,2,0,0><<<gemm_grid(M,N),256,0,stream>>>(A,lda,W,ldb,bias,(void*)O,ldc,M,N,K,0,1,0);
    };
    auto g_f32 = [&](const h16* A, int lda, const h16* W, int ldb,
                     float* O, int ldc, int M, int N, int K) {
        gemm_wmma<0,0,0,1><<<gemm_grid(M,N),256,0,stream>>>(A,lda,W,ldb,nullptr,(void*)O,ldc,M,N,K,0,1,0);
    };
    auto g_resid = [&](const h16* A, int lda, const h16* W, int ldb, const float* bias,
                       float* O, int ldc, int M, int N, int K) {
        gemm_wmma<0,0,0,3><<<gemm_grid(M,N),256,0,stream>>>(A,lda,W,ldb,bias,(void*)O,ldc,M,N,K,0,1,0);
    };
    auto g_f16T = [&](const h16* A, int lda, const h16* W, int ldb,
                      h16* O, int ldc, int M, int N, int K) {
        gemm_wmma<0,0,0,4><<<gemm_grid(M,N),256,0,stream>>>(A,lda,W,ldb,nullptr,(void*)O,ldc,M,N,K,0,1,0);
    };

    // ===================== 1) DNA conv (im2col) + pool =====================
    k_dna_im2col<<<ew_grid(L0*64),256,0,stream>>>(X, IM2, L0);
    k_wt_dna<<<ew_grid(512*64),256,0,stream>>>(dna_w, CW);
    gemm_wmma<0,0,1,0><<<gemm_grid(L0,512),256,0,stream>>>(IM2,64, CW,64, dna_b,
        (void*)A0, 512, (int)L0, 512, 64, 0,1,0);

    // ===================== 2) Residual conv tower =====================
    const int rt_cin [5] = {512, 608, 736, 896, 1056};
    const int rt_cout[5] = {608, 736, 896, 1056, 1280};
    h16* cur = A0;
    long Lc = L1;
    for (int i = 0; i < 5; i++) {
        int cin = rt_cin[i], cout = rt_cout[i], K = cin * 5;
        const float* g  = in_f(RT + 4*i + 0);
        const float* b  = in_f(RT + 4*i + 1);
        const float* w  = in_f(RT + 4*i + 2);
        const float* wb = in_f(RT + 4*i + 3);
        k_bngelu<<<ew_grid(Lc*cin),256,0,stream>>>(cur, g, b, PRE, Lc*cin, cin);
        k_wt_conv<<<ew_grid((long)cout*cin*5),256,0,stream>>>(w, CW, cout, cin, 5);
        h16* outb = (cur == A0) ? C1 : A0;
        if (i < 4) {
            gemm_wmma<1,0,1,0><<<gemm_grid(Lc,cout),256,0,stream>>>(PRE,0, CW,K, wb,
                (void*)outb, cout, (int)Lc, cout, K, cin, 5, 2);
            Lc >>= 1;
        } else {
            gemm_wmma<1,0,0,0><<<gemm_grid(Lc,cout),256,0,stream>>>(PRE,0, CW,K, wb,
                (void*)outb, cout, (int)Lc, cout, K, cin, 5, 2);
        }
        cur = outb;
    }
    // cur == C1 == x_u0 raw [16384][1280]

    // ===================== 3) unet1 / hc0 / hc1 =====================
    h16* P8 = A0;
    k_maxpool<<<ew_grid((long)8192*1280),256,0,stream>>>(C1, P8, (long)8192*1280, 1280);
    k_bngelu<<<ew_grid((long)8192*1280),256,0,stream>>>(P8, in_f(UN+0), in_f(UN+1), PRE, (long)8192*1280, 1280);
    k_wt_conv<<<ew_grid((long)1536*1280*5),256,0,stream>>>(in_f(UN+2), CW, 1536, 1280, 5);
    gemm_wmma<1,0,0,0><<<gemm_grid(8192,1536),256,0,stream>>>(PRE,0, CW,6400, in_f(UN+3),
        (void*)XU1, 1536, 8192, 1536, 6400, 1280, 5, 2);

    // hc0: k1 conv on raw x_u0
    k_bngelu<<<ew_grid((long)16384*1280),256,0,stream>>>(C1, in_f(HC0+0), in_f(HC0+1), PRE, (long)16384*1280, 1280);
    k_cvt_f32_f16<<<ew_grid((long)1536*1280),256,0,stream>>>(in_f(HC0+2), CW, (long)1536*1280);
    g_f16(PRE,1280, CW,1280, in_f(HC0+3), XU0C,1536, 16384,1536,1280);

    // hc1: k1 conv on raw x_u1
    k_bngelu<<<ew_grid((long)8192*1536),256,0,stream>>>(XU1, in_f(HC1+0), in_f(HC1+1), PRE, (long)8192*1536, 1536);
    k_cvt_f32_f16<<<ew_grid((long)1536*1536),256,0,stream>>>(in_f(HC1+2), CW, (long)1536*1536);
    g_f16(PRE,1536, CW,1536, in_f(HC1+3), XU1C,1536, 8192,1536,1536);

    // transformer input: pool(x_u1) -> fp32 residual stream
    k_maxpool<<<ew_grid((long)4096*1536),256,0,stream>>>(XU1, X4, (long)4096*1536, 1536);
    k_cvt_f16_f32<<<ew_grid((long)4096*1536),256,0,stream>>>(X4, T, (long)4096*1536);
    k_pe<<<32,256,0,stream>>>(PEB, 4096);

    // ===================== 4) Transformer (8 layers) =====================
    for (int l = 0; l < 8; l++) {
        int p = LB + 16 * l;
        const float* ln1g = in_f(p+0); const float* ln1b = in_f(p+1);
        const float* wq = in_f(p+2);  const float* wk = in_f(p+3);
        const float* wv = in_f(p+4);  const float* wo = in_f(p+5);
        const float* bo = in_f(p+6);  const float* wrk = in_f(p+7);
        const float* rcb = in_f(p+8); const float* rpb = in_f(p+9);
        const float* ln2g = in_f(p+10); const float* ln2b = in_f(p+11);
        const float* w1 = in_f(p+12); const float* b1 = in_f(p+13);
        const float* w2 = in_f(p+14); const float* b2 = in_f(p+15);

        k_ln<<<4096,256,0,stream>>>(T, ln1g, ln1b, Y, 1536);

        // rel_k = pe @ wrk  [8191][512]
        k_wt_dense<<<ew_grid((long)32*512),256,0,stream>>>(wrk, WS, 32, 512);
        g_f16(PEB,32, WS,32, nullptr, RELK,512, 8191,512,32);

        // q (fp32), k, v^T
        k_wt_dense<<<ew_grid((long)1536*512),256,0,stream>>>(wq, WS, 1536, 512);
        g_f32(Y,1536, WS,1536, QF,512, 4096,512,1536);
        k_qcqp<<<ew_grid((long)4096*512),256,0,stream>>>(QF, rcb, rpb, QC, QP, (long)4096*512);
        k_wt_dense<<<ew_grid((long)1536*512),256,0,stream>>>(wk, WS, 1536, 512);
        g_f16(Y,1536, WS,1536, nullptr, KB,512, 4096,512,1536);
        k_wt_dense<<<ew_grid((long)1536*1536),256,0,stream>>>(wv, WS, 1536, 1536);
        g_f16T(Y,1536, WS,1536, VT,4096, 4096,1536,1536);

        for (int h = 0; h < 8; h++) {
            // content scores + full relative matrix, then fused shift+softmax
            g_f32(QC + h*64,512, KB + h*64,512, Sbuf,4096, 4096,4096,64);
            g_f32(QP + h*64,512, RELK + h*64,512, Rbuf,8192, 4096,8191,64);
            k_softmax_rel<<<4096,256,0,stream>>>(Sbuf, Rbuf, Pbuf, 4096, 8192);
            g_f16(Pbuf,4096, VT + (long)h*192*4096,4096, nullptr,
                  ATTN + h*192,1536, 4096,192,4096);
        }

        k_wt_dense<<<ew_grid((long)1536*1536),256,0,stream>>>(wo, WS, 1536, 1536);
        g_resid(ATTN,1536, WS,1536, bo, T,1536, 4096,1536,1536);

        k_ln<<<4096,256,0,stream>>>(T, ln2g, ln2b, Y, 1536);
        k_wt_dense<<<ew_grid((long)1536*3072),256,0,stream>>>(w1, WS, 1536, 3072);
        g_relu(Y,1536, WS,1536, b1, FF1,3072, 4096,3072,1536);
        k_wt_dense<<<ew_grid((long)3072*1536),256,0,stream>>>(w2, WS, 3072, 1536);
        g_resid(FF1,3072, WS,3072, b2, T,1536, 4096,1536,3072);
    }

    // ===================== 5) Upsampling head =====================
    k_cvt_f32_f16<<<ew_grid((long)4096*1536),256,0,stream>>>(T, XT, (long)4096*1536);

    // up1 + repeat + add x_u1c
    k_bngelu<<<ew_grid((long)4096*1536),256,0,stream>>>(XT, in_f(UP1+0), in_f(UP1+1), Y, (long)4096*1536, 1536);
    k_cvt_f32_f16<<<ew_grid((long)1536*1536),256,0,stream>>>(in_f(UP1+2), CW, (long)1536*1536);
    g_f16(Y,1536, CW,1536, in_f(UP1+3), U1B,1536, 4096,1536,1536);
    k_repeat_add<<<ew_grid((long)8192*1536),256,0,stream>>>(U1B, XU1C, X8, (long)8192*1536, 1536);

    // sep1: depthwise k3 + pointwise
    k_dwconv3<<<ew_grid((long)8192*1536),256,0,stream>>>(X8, in_f(SEP1+0), D8, 8192, 1536);
    k_cvt_f32_f16<<<ew_grid((long)1536*1536),256,0,stream>>>(in_f(SEP1+1), CW, (long)1536*1536);
    g_f16(D8,1536, CW,1536, in_f(SEP1+2), S8,1536, 8192,1536,1536);

    // up0 + repeat + add x_u0c
    k_bngelu<<<ew_grid((long)8192*1536),256,0,stream>>>(S8, in_f(UP0+0), in_f(UP0+1), D8, (long)8192*1536, 1536);
    k_cvt_f32_f16<<<ew_grid((long)1536*1536),256,0,stream>>>(in_f(UP0+2), CW, (long)1536*1536);
    g_f16(D8,1536, CW,1536, in_f(UP0+3), U8,1536, 8192,1536,1536);
    k_repeat_add<<<ew_grid((long)16384*1536),256,0,stream>>>(U8, XU0C, X16, (long)16384*1536, 1536);

    // sep0
    k_dwconv3<<<ew_grid((long)16384*1536),256,0,stream>>>(X16, in_f(SEP0+0), D16, 16384, 1536);
    k_cvt_f32_f16<<<ew_grid((long)1536*1536),256,0,stream>>>(in_f(SEP0+1), CW, (long)1536*1536);
    g_f16(D16,1536, CW,1536, in_f(SEP0+2), S16,1536, 16384,1536,1536);

    // ===================== 6) Crop + final cblock + gelu -> d_out =====================
    h16* PREF = (h16*)PRE;
    k_bngelu<<<ew_grid((long)16384*1536),256,0,stream>>>(S16, in_f(FIN+0), in_f(FIN+1), PREF, (long)16384*1536, 1536);
    k_cvt_f32_f16<<<ew_grid((long)1920*1536),256,0,stream>>>(in_f(FIN+2), CW, (long)1920*1536);
    // crop 16 rows each side; output fp32 channel-major [1920][16352]
    gemm_wmma<0,1,0,2><<<gemm_grid(16352,1920),256,0,stream>>>(
        PREF + (long)16*1536, 1536, CW, 1536, in_f(FIN+3),
        d_out, 16352, 16352, 1920, 1536, 0,1,0);
}
